// TransformerBlock_8177617731890
// MI455X (gfx1250) — matrix operand order repaired and checked by reading, not compiled, not run on hardware
//
#include <hip/hip_runtime.h>
#include <stdint.h>

#define BB   2
#define TT   2048
#define CC   2048
#define HH   16
#define HD   128
#define MLPD 8192
#define BT   (BB * TT)
#define EPSF 1e-5f

typedef __bf16 bf16x16 __attribute__((ext_vector_type(16)));
typedef float  f32x8   __attribute__((ext_vector_type(8)));
typedef unsigned short u16x8 __attribute__((ext_vector_type(8)));
typedef unsigned int u32x4v __attribute__((ext_vector_type(4)));
typedef int i32x8v __attribute__((ext_vector_type(8)));
typedef int i32x4v __attribute__((ext_vector_type(4)));

union Frag16 {
    bf16x16 v;
    u16x8   h[2];
    unsigned short u[16];
};
union F8 {
    f32x8 v;
    float f[8];
};

__device__ __forceinline__ f32x8 zero8() {
    f32x8 z = {0.f, 0.f, 0.f, 0.f, 0.f, 0.f, 0.f, 0.f};
    return z;
}

// fp32 -> bf16 (round-to-nearest-even), raw u16 storage
__device__ __forceinline__ unsigned short f2bf(float x) {
    union { float f; unsigned int u; } c;
    c.f = x;
    unsigned int r = c.u + 0x7FFFu + ((c.u >> 16) & 1u);
    return (unsigned short)(r >> 16);
}

// Raw LDS byte offset of a __shared__ object (generic LDS ptr low 32 bits)
__device__ __forceinline__ unsigned lds_off(const void* p) {
    return (unsigned)(uintptr_t)p;
}

// ---------------------------------------------------------------------------
// TDM: DMA one 128-row x 32-col bf16 tile (row-major, row stride Kdim) from
// global memory into LDS with 8 pad halves after each 32-half row
// (pad_interval code 3 = 16 DWORDs, pad_amount code 3 = 4 DWORDs -> stride 40).
// D# layout per CDNA5 ISA ch.8 (group0 128b, group1 256b, groups 2/3 unused).
// This toolchain exposes the 6-arg builtin (extra int32x8 group before cpol).
// ---------------------------------------------------------------------------
__device__ __forceinline__ void tdm_tile_load(unsigned ldsByteOff,
                                              const unsigned short* gtile,
                                              int Kdim, int rowsTotal) {
    unsigned long long ga = (unsigned long long)(uintptr_t)gtile;
    u32x4v g0;
    g0[0] = 1u;                                   // count=1, user descriptor
    g0[1] = ldsByteOff;                           // lds_addr (bytes)
    g0[2] = (unsigned)(ga & 0xffffffffu);         // global_addr[31:0]
    g0[3] = (unsigned)((ga >> 32) & 0x01ffffffu)  // global_addr[56:32]
            | (2u << 30);                         // type = 2 ("image")
    i32x8v g1;
    g1[0] = (int)(0x00010000u      // data_size=1 (2 bytes)
                  | (1u << 20)     // pad_enable
                  | (3u << 22)     // pad_interval: 16 DWORDs stored
                  | (3u << 25));   // pad_amount: 4 DWORDs (8 halves)
    g1[1] = (int)(((unsigned)Kdim & 0xffffu) << 16);        // tensor_dim0[15:0]
    g1[2] = (int)((((unsigned)Kdim >> 16) & 0xffffu)        // tensor_dim0[31:16]
                  | (((unsigned)rowsTotal & 0xffffu) << 16)); // tensor_dim1[15:0]
    g1[3] = (int)(((((unsigned)rowsTotal) >> 16) & 0xffffu) // tensor_dim1[31:16]
                  | (32u << 16));                           // tile_dim0 = 32
    g1[4] = 128;                                            // tile_dim1 = 128
    g1[5] = Kdim;                                           // tensor_dim0_stride lo
    g1[6] = 0;
    g1[7] = 0;
    i32x4v z4 = {0, 0, 0, 0};
    i32x8v z8 = {0, 0, 0, 0, 0, 0, 0, 0};
    __builtin_amdgcn_tensor_load_to_lds(g0, g1, z4, z4, z8, 0);
}

// ---------------------------------------------------------------------------
// Elementwise fp32 -> bf16 conversion (weights)
// ---------------------------------------------------------------------------
__global__ void cvt_f32_bf16(const float* __restrict__ in,
                             unsigned short* __restrict__ out, int n) {
    int i = blockIdx.x * blockDim.x + threadIdx.x;
    if (i < n) out[i] = f2bf(in[i]);
}

// ---------------------------------------------------------------------------
// RMSNorm (one block per row of length CC) fused with bf16 conversion
// ---------------------------------------------------------------------------
__global__ __launch_bounds__(256) void rmsnorm_bf16(const float* __restrict__ x,
                                                    unsigned short* __restrict__ h) {
    __shared__ float red[256];
    const int row = blockIdx.x;
    const float* xr = x + (size_t)row * CC;
    float s = 0.f;
    for (int c = threadIdx.x; c < CC; c += 256) {
        float v = xr[c];
        s += v * v;
    }
    red[threadIdx.x] = s;
    __syncthreads();
    for (int off = 128; off > 0; off >>= 1) {
        if (threadIdx.x < off) red[threadIdx.x] += red[threadIdx.x + off];
        __syncthreads();
    }
    const float inv = rsqrtf(red[0] / (float)CC + EPSF);
    unsigned short* hr = h + (size_t)row * CC;
    for (int c = threadIdx.x; c < CC; c += 256) hr[c] = f2bf(xr[c] * inv);
}

// ---------------------------------------------------------------------------
// NT GEMM: C[M,N] = A[M,K] * B[N,K]^T   (A,B bf16 row-major, K-major rows)
// Block tile 128x128x32, 256 threads = 8 waves, wave tile 64x32 (4x2 WMMAs).
// LDS tiles double-buffered and filled by the Tensor Data Mover (wave 0 issues
// the next tile's DMA while all waves compute on the current buffer).
// EPI: 0 = fp32 out, 1 = fp32 out + residual, 2 = relu^2 -> bf16, 3 = bf16
// ---------------------------------------------------------------------------
template <int EPI>
__global__ void __launch_bounds__(256) gemm_nt(const unsigned short* __restrict__ A,
                                               const unsigned short* __restrict__ Bm,
                                               const float* __restrict__ resid,
                                               float* __restrict__ Cf,
                                               unsigned short* __restrict__ Cb,
                                               int M, int N, int K) {
    __shared__ __align__(16) unsigned short As[2][128 * 40];  // stride 40 (TDM pad)
    __shared__ __align__(16) unsigned short Bs[2][128 * 40];

    const int tid  = threadIdx.x;
    const int lane = tid & 31;
    const int wave = tid >> 5;
    const int wm   = (wave >> 2) * 64;   // wave M offset (2 waves along M)
    const int wn   = (wave & 3) * 32;    // wave N offset (4 waves along N)
    const int bm   = blockIdx.y * 128;
    const int bn   = blockIdx.x * 128;
    const int m15  = lane & 15;
    const int hf   = lane >> 4;

    F8 acc[4][2];
#pragma unroll
    for (int i = 0; i < 4; i++)
#pragma unroll
        for (int j = 0; j < 2; j++) acc[i][j].v = zero8();

    // Prologue: DMA first tiles into buffer 0.
    if (tid < 32) {
        tdm_tile_load(lds_off(&As[0][0]), A  + (size_t)bm * K, K, M);
        tdm_tile_load(lds_off(&Bs[0][0]), Bm + (size_t)bn * K, K, N);
        __builtin_amdgcn_s_wait_tensorcnt(0);
    }
    __syncthreads();

    int buf = 0;
    for (int kb = 0; kb < K; kb += 32) {
        const bool more = (kb + 32) < K;
        if (more && tid < 32) {
            tdm_tile_load(lds_off(&As[buf ^ 1][0]),
                          A + (size_t)bm * K + kb + 32, K, M);
            tdm_tile_load(lds_off(&Bs[buf ^ 1][0]),
                          Bm + (size_t)bn * K + kb + 32, K, N);
        }

        Frag16 af[4], bf[2];
#pragma unroll
        for (int tm = 0; tm < 4; tm++) {
            // A fragment: lane row = m15, K chunks at hf*8 and hf*8+16
            const unsigned short* p = &As[buf][(wm + tm * 16 + m15) * 40 + hf * 8];
            af[tm].h[0] = *(const u16x8*)p;
            af[tm].h[1] = *(const u16x8*)(p + 16);
        }
#pragma unroll
        for (int tn = 0; tn < 2; tn++) {
            // B fragment: lane col = m15, K chunks at hf*8 and hf*8+16
            const unsigned short* p = &Bs[buf][(wn + tn * 16 + m15) * 40 + hf * 8];
            bf[tn].h[0] = *(const u16x8*)p;
            bf[tn].h[1] = *(const u16x8*)(p + 16);
        }
#pragma unroll
        for (int tm = 0; tm < 4; tm++)
#pragma unroll
            for (int tn = 0; tn < 2; tn++)
                acc[tm][tn].v = __builtin_amdgcn_wmma_f32_16x16x32_bf16(
                    false, af[tm].v, false, bf[tn].v, (short)0, acc[tm][tn].v,
                    false, false);

        if (more && tid < 32) __builtin_amdgcn_s_wait_tensorcnt(0);
        __syncthreads();
        buf ^= 1;
    }

    // Epilogue: C element (vgpr j, lane) -> row j + 8*hf, col m15
#pragma unroll
    for (int tm = 0; tm < 4; tm++)
#pragma unroll
        for (int tn = 0; tn < 2; tn++)
#pragma unroll
            for (int j = 0; j < 8; j++) {
                const int row = bm + wm + tm * 16 + j + hf * 8;
                const int col = bn + wn + tn * 16 + m15;
                const size_t idx = (size_t)row * N + col;
                float v = acc[tm][tn].f[j];
                if (EPI == 0) {
                    Cf[idx] = v;
                } else if (EPI == 1) {
                    Cf[idx] = resid[idx] + v;
                } else if (EPI == 2) {
                    float r = v > 0.f ? v : 0.f;
                    Cb[idx] = f2bf(r * r);
                } else {
                    Cb[idx] = f2bf(v);
                }
            }
}

// ---------------------------------------------------------------------------
// Causal flash attention. One wave per 16-row Q tile, K/V tiles of 32.
// q,k,v,o are [BT, CC] bf16 with head h at columns h*HD .. h*HD+127.
// ---------------------------------------------------------------------------
__global__ void __launch_bounds__(128) flash_attn(const unsigned short* __restrict__ q,
                                                  const unsigned short* __restrict__ k,
                                                  const unsigned short* __restrict__ v,
                                                  unsigned short* __restrict__ o) {
    __shared__ __align__(16) unsigned short Ps[4][16 * 32];  // per-wave P tile

    const int lane = threadIdx.x & 31;
    const int wv   = threadIdx.x >> 5;
    const int gw   = blockIdx.x * 4 + wv;
    const int bh   = gw >> 7;          // 128 q-tiles per (b,h)
    const int qt   = gw & 127;
    const int b    = bh >> 4;
    const int h    = bh & 15;
    const size_t rowbase = (size_t)b * TT;
    const int hoff = h * HD;
    const int qr   = qt * 16;
    const int m15  = lane & 15;
    const int hf   = lane >> 4;
    const float scale = 0.08838834764831845f;  // 1/sqrt(128)

    // Q fragments (A layout): row = m15, K chunks hf*8 and hf*8+16 per 32-chunk
    Frag16 Qf[4];
    {
        const unsigned short* qp =
            q + (rowbase + qr + m15) * CC + hoff + hf * 8;
#pragma unroll
        for (int dc = 0; dc < 4; dc++) {
            Qf[dc].h[0] = *(const u16x8*)(qp + dc * 32);
            Qf[dc].h[1] = *(const u16x8*)(qp + dc * 32 + 16);
        }
    }

    F8 O[8];
#pragma unroll
    for (int c = 0; c < 8; c++) O[c].v = zero8();
    float Mr[8], Lr[8];
#pragma unroll
    for (int j = 0; j < 8; j++) { Mr[j] = -1e30f; Lr[j] = 0.f; }

    const int nk = (qr + 47) >> 5;  // causal #32-wide K tiles
    for (int kt = 0; kt < nk; kt++) {
        const int kb = kt * 32;

        // Prefetch next K/V tile rows into cache (global_prefetch_b8)
        if (kt + 1 < nk) {
            __builtin_prefetch(k + (rowbase + kb + 32 + m15) * CC + hoff, 0, 1);
            __builtin_prefetch(v + (rowbase + kb + 32 + m15) * CC + hoff, 0, 1);
        }

        // S = Q K^T  (two 16-col sub-tiles); batch all 8 fragment loads first
        F8 S[2];
#pragma unroll
        for (int ns = 0; ns < 2; ns++) {
            const unsigned short* kp =
                k + (rowbase + kb + ns * 16 + m15) * CC + hoff + hf * 8;
            Frag16 Kf[4];
#pragma unroll
            for (int dc = 0; dc < 4; dc++) {
                Kf[dc].h[0] = *(const u16x8*)(kp + dc * 32);
                Kf[dc].h[1] = *(const u16x8*)(kp + dc * 32 + 16);
            }
            f32x8 sacc = zero8();
#pragma unroll
            for (int dc = 0; dc < 4; dc++)
                sacc = __builtin_amdgcn_wmma_f32_16x16x32_bf16(
                    false, Qf[dc].v, false, Kf[dc].v, (short)0, sacc, false, false);
            S[ns].v = sacc;
        }

        // scale + causal mask (element (j,lane): row qr+j+8*hf, col kb+ns*16+m15)
#pragma unroll
        for (int ns = 0; ns < 2; ns++)
#pragma unroll
            for (int j = 0; j < 8; j++) {
                const int r = qr + j + hf * 8;
                const int c = kb + ns * 16 + m15;
                float sv = S[ns].f[j] * scale;
                S[ns].f[j] = (c > r) ? -1e30f : sv;
            }

        // row max across the 16 lanes of this half (cols live across lanes)
        float nm[8];
#pragma unroll
        for (int j = 0; j < 8; j++) nm[j] = fmaxf(S[0].f[j], S[1].f[j]);
#pragma unroll
        for (int msk = 1; msk < 16; msk <<= 1)
#pragma unroll
            for (int j = 0; j < 8; j++)
                nm[j] = fmaxf(nm[j], __shfl_xor(nm[j], msk, 16));

        float alpha[8];
#pragma unroll
        for (int j = 0; j < 8; j++) {
            float mn = fmaxf(Mr[j], nm[j]);
            alpha[j] = __expf(Mr[j] - mn);
            Mr[j] = mn;
        }

        // P = exp(S - M); write to LDS in row-major [16][32] for A-layout reload
        float ls[8];
#pragma unroll
        for (int j = 0; j < 8; j++) ls[j] = 0.f;
#pragma unroll
        for (int ns = 0; ns < 2; ns++)
#pragma unroll
            for (int j = 0; j < 8; j++) {
                float p = __expf(S[ns].f[j] - Mr[j]);
                ls[j] += p;
                Ps[wv][(j + hf * 8) * 32 + ns * 16 + m15] = f2bf(p);
            }
#pragma unroll
        for (int msk = 1; msk < 16; msk <<= 1)
#pragma unroll
            for (int j = 0; j < 8; j++) ls[j] += __shfl_xor(ls[j], msk, 16);
#pragma unroll
        for (int j = 0; j < 8; j++) Lr[j] = Lr[j] * alpha[j] + ls[j];
#pragma unroll
        for (int c = 0; c < 8; c++)
#pragma unroll
            for (int j = 0; j < 8; j++) O[c].f[j] *= alpha[j];

        // cross-lane LDS transpose: wait for all DS stores of this wave
        asm volatile("s_wait_dscnt 0" ::: "memory");

        // reload P as A fragment (16x32)
        Frag16 Pf;
        {
            const unsigned short* pp = &Ps[wv][m15 * 32 + hf * 16];
            Pf.h[0] = *(const u16x8*)pp;
            Pf.h[1] = *(const u16x8*)(pp + 8);
        }

        // O += P @ V over 8 column sub-tiles of HD
#pragma unroll
        for (int c = 0; c < 8; c++) {
            Frag16 Vf;
            const unsigned short* vp =
                v + (rowbase + kb + hf * 16) * CC + hoff + c * 16 + m15;
#pragma unroll
            for (int jj = 0; jj < 16; jj++) Vf.u[jj] = vp[(size_t)jj * CC];
            O[c].v = __builtin_amdgcn_wmma_f32_16x16x32_bf16(
                false, Pf.v, false, Vf.v, (short)0, O[c].v, false, false);
        }
    }

    float inv[8];
#pragma unroll
    for (int j = 0; j < 8; j++) inv[j] = 1.0f / Lr[j];
#pragma unroll
    for (int c = 0; c < 8; c++)
#pragma unroll
        for (int j = 0; j < 8; j++) {
            const size_t idx =
                (rowbase + qr + j + hf * 8) * CC + hoff + c * 16 + m15;
            o[idx] = f2bf(O[c].f[j] * inv[j]);
        }
}

// ---------------------------------------------------------------------------
// Host orchestration
// ---------------------------------------------------------------------------
extern "C" void kernel_launch(void* const* d_in, const int* in_sizes, int n_in,
                              void* d_out, int out_size, void* d_ws, size_t ws_size,
                              hipStream_t stream) {
    (void)in_sizes; (void)n_in; (void)out_size; (void)ws_size;
    const float* x   = (const float*)d_in[0];
    const float* wq  = (const float*)d_in[1];
    const float* wk  = (const float*)d_in[2];
    const float* wv  = (const float*)d_in[3];
    const float* wo  = (const float*)d_in[4];
    const float* fc1 = (const float*)d_in[5];
    const float* fc2 = (const float*)d_in[6];
    float* out = (float*)d_out;

    char* ws = (char*)d_ws;
    size_t off = 0;
    auto alloc = [&](size_t bytes) {
        char* p = ws + off;
        off += (bytes + 255) & ~(size_t)255;
        return p;
    };
    unsigned short* wqb  = (unsigned short*)alloc((size_t)CC * CC * 2);
    unsigned short* wkb  = (unsigned short*)alloc((size_t)CC * CC * 2);
    unsigned short* wvb  = (unsigned short*)alloc((size_t)CC * CC * 2);
    unsigned short* wob  = (unsigned short*)alloc((size_t)CC * CC * 2);
    unsigned short* fc1b = (unsigned short*)alloc((size_t)MLPD * CC * 2);
    unsigned short* fc2b = (unsigned short*)alloc((size_t)CC * MLPD * 2);
    unsigned short* hb   = (unsigned short*)alloc((size_t)BT * CC * 2);
    unsigned short* qb   = (unsigned short*)alloc((size_t)BT * CC * 2);
    unsigned short* kb   = (unsigned short*)alloc((size_t)BT * CC * 2);
    unsigned short* vb   = (unsigned short*)alloc((size_t)BT * CC * 2);
    unsigned short* attb = (unsigned short*)alloc((size_t)BT * CC * 2);
    float*          x1   = (float*)alloc((size_t)BT * CC * 4);
    unsigned short* h2b  = (unsigned short*)alloc((size_t)BT * CC * 2);
    unsigned short* mb   = (unsigned short*)alloc((size_t)BT * MLPD * 2);

    const int nW  = CC * CC;      // 4 Mi
    const int nFC = MLPD * CC;    // 16 Mi
    cvt_f32_bf16<<<(nW + 255) / 256, 256, 0, stream>>>(wq, wqb, nW);
    cvt_f32_bf16<<<(nW + 255) / 256, 256, 0, stream>>>(wk, wkb, nW);
    cvt_f32_bf16<<<(nW + 255) / 256, 256, 0, stream>>>(wv, wvb, nW);
    cvt_f32_bf16<<<(nW + 255) / 256, 256, 0, stream>>>(wo, wob, nW);
    cvt_f32_bf16<<<(nFC + 255) / 256, 256, 0, stream>>>(fc1, fc1b, nFC);
    cvt_f32_bf16<<<(nFC + 255) / 256, 256, 0, stream>>>(fc2, fc2b, nFC);

    // h = rmsnorm(x) in bf16
    rmsnorm_bf16<<<BT, 256, 0, stream>>>(x, hb);

    // q/k/v = h @ W^T  (bf16 out)
    dim3 gQKV(CC / 128, BT / 128);
    gemm_nt<3><<<gQKV, 256, 0, stream>>>(hb, wqb, nullptr, nullptr, qb, BT, CC, CC);
    gemm_nt<3><<<gQKV, 256, 0, stream>>>(hb, wkb, nullptr, nullptr, kb, BT, CC, CC);
    gemm_nt<3><<<gQKV, 256, 0, stream>>>(hb, wvb, nullptr, nullptr, vb, BT, CC, CC);

    // causal flash attention: 2*16*(2048/16) = 4096 waves, 4 waves/block
    flash_attn<<<(BB * HH * (TT / 16)) / 4, 128, 0, stream>>>(qb, kb, vb, attb);

    // x1 = x + att @ wo^T  (fp32)
    gemm_nt<1><<<gQKV, 256, 0, stream>>>(attb, wob, x, x1, nullptr, BT, CC, CC);

    // h2 = rmsnorm(x1)
    rmsnorm_bf16<<<BT, 256, 0, stream>>>(x1, h2b);

    // m = relu(h2 @ fc1^T)^2  (bf16)
    dim3 gFC1(MLPD / 128, BT / 128);
    gemm_nt<2><<<gFC1, 256, 0, stream>>>(h2b, fc1b, nullptr, nullptr, mb, BT, MLPD, CC);

    // out = x1 + m @ fc2^T  (fp32)
    dim3 gFC2(CC / 128, BT / 128);
    gemm_nt<1><<<gFC2, 256, 0, stream>>>(mb, fc2b, x1, out, nullptr, BT, CC, MLPD);
}
